// GNN_79044578115825
// MI455X (gfx1250) — matrix operand order repaired and checked by reading, not compiled, not run on hardware
//
#include <hip/hip_runtime.h>
#include <hip/hip_bf16.h>

// ---------------------------------------------------------------------------
// GNN (2x GCNConv + mean-pool + projection) for MI455X / gfx1250, wave32.
// Dense GEMMs use v_wmma_f32_16x16x32_f16 (f16 A/B, fp32 accum).
// Scatter-aggregation uses per-molecule LDS fp32 atomics (ds_add_f32).
// ---------------------------------------------------------------------------

#define BATCH   256
#define NATOM   1024
#define NEDGE   8192
#define NEG_SLOPE 0.01f

typedef __attribute__((ext_vector_type(16))) _Float16 v16h;
typedef __attribute__((ext_vector_type(8)))  float    v8f;

union AV { uint4 q[2]; v16h v; };

__device__ __forceinline__ float leaky(float v) {
    return v >= 0.0f ? v : NEG_SLOPE * v;
}

// ---------------- workspace layout (byte offsets, 16B aligned) -------------
#define OFF_DEG    (size_t)0                         // 1024 u32      (4 KB)
#define OFF_DINV   (size_t)4096                      // 1024 f32      (4 KB)
#define OFF_NSELF  (size_t)8192                      // 1024 f32      (4 KB)
#define OFF_NORME  (size_t)12288                     // 8192 f32      (32 KB)
#define OFF_W2H    (size_t)45056                     // 64*64 f16     (8 KB)
#define OFF_WPH    (size_t)53248                     // 64*128 f16    (16 KB)
#define OFF_GH     (size_t)69632                     // 256*64 f16    (32 KB)
#define OFF_H1H    (size_t)102400                    // B*N*64 f16    (32 MB)
#define OFF_HW2    (size_t)(102400 + (size_t)BATCH*NATOM*64*2)  // B*N*64 f32 (64 MB)

// ---------------------------------------------------------------------------
// Kernel 0: zero degree counters (deterministic re-init every call)
__global__ void k_zero_deg(unsigned int* deg) {
    deg[blockIdx.x * 256 + threadIdx.x] = 0u;
}

// Kernel 1: in-degree via global atomics (8192 edges)
__global__ void k_count_deg(const int* __restrict__ ei, unsigned int* deg) {
    int e = blockIdx.x * 256 + threadIdx.x;
    atomicAdd(&deg[ei[NEDGE + e]], 1u);
}

// Kernel 2: dinv = rsqrt(1+deg), norm_self = dinv*dinv
__global__ void k_norm_node(const unsigned int* __restrict__ deg,
                            float* __restrict__ dinv, float* __restrict__ nself) {
    int i = blockIdx.x * 256 + threadIdx.x;
    float d  = 1.0f + (float)deg[i];
    float di = rsqrtf(d);
    dinv[i]  = di;
    nself[i] = di * di;
}

// Kernel 3: per-edge norm
__global__ void k_norm_edge(const int* __restrict__ ei,
                            const float* __restrict__ dinv,
                            float* __restrict__ norme) {
    int e = blockIdx.x * 256 + threadIdx.x;
    norme[e] = dinv[ei[e]] * dinv[ei[NEDGE + e]];
}

// Kernel 4: convert W2 (64x64) and Wp (64x128) fp32 -> f16 for WMMA B operands
__global__ void k_cvt_weights(const float* __restrict__ W2,
                              const float* __restrict__ Wp,
                              _Float16* __restrict__ w2h,
                              _Float16* __restrict__ wph) {
    int i = blockIdx.x * 256 + threadIdx.x;       // 12288 total
    if (i < 64 * 64)      w2h[i]          = (_Float16)W2[i];
    else                  wph[i - 64*64]  = (_Float16)Wp[i - 64*64];
}

// ---------------------------------------------------------------------------
// Kernel 5: GCN layer 1, fused per molecule.
// hw1 = x @ W1 is only 3 MACs/feature -> recompute per edge instead of
// materializing. Aggregate in 1024x16 fp32 LDS chunks (4 chunks of 16 feats).
// Output h1 in f16 (WMMA A operand precision for layer 2).
__global__ __launch_bounds__(256) void k_gcn1(
    const float* __restrict__ x, const int* __restrict__ ei,
    const float* __restrict__ W1, const float* __restrict__ b1,
    const float* __restrict__ nself, const float* __restrict__ norme,
    _Float16* __restrict__ h1h)
{
    __shared__ float xs[NATOM * 3];      // 12 KB
    __shared__ float agg[NATOM * 16];    // 64 KB
    __shared__ float w1s[3 * 64];        // 768 B

    const int b   = blockIdx.x;
    const int tid = threadIdx.x;
    const float* xb = x + (size_t)b * NATOM * 3;

    for (int i = tid; i < NATOM * 3; i += 256) xs[i] = xb[i];
    if (tid < 192) w1s[tid] = W1[tid];
    __syncthreads();

    for (int c = 0; c < 4; ++c) {
        for (int i = tid; i < NATOM * 16; i += 256) agg[i] = 0.0f;
        __syncthreads();

        // edge scatter: agg[dst] += norm_e * (x[src] @ W1)[chunk]
        for (int e = tid; e < NEDGE; e += 256) {
            int   s  = ei[e];
            int   d  = ei[NEDGE + e];
            float w  = norme[e];
            float x0 = xs[s * 3 + 0], x1 = xs[s * 3 + 1], x2 = xs[s * 3 + 2];
#pragma unroll
            for (int f = 0; f < 16; ++f) {
                int j = c * 16 + f;
                float hw = x0 * w1s[j] + x1 * w1s[64 + j] + x2 * w1s[128 + j];
                atomicAdd(&agg[d * 16 + f], w * hw);   // ds_add_f32
            }
        }
        __syncthreads();

        // self term + bias + LeakyReLU, emit f16
        for (int n = tid; n < NATOM; n += 256) {
            float x0 = xs[n * 3 + 0], x1 = xs[n * 3 + 1], x2 = xs[n * 3 + 2];
            float ns = nself[n];
            _Float16* orow = h1h + ((size_t)b * NATOM + n) * 64;
#pragma unroll
            for (int f = 0; f < 16; ++f) {
                int j = c * 16 + f;
                float hw = x0 * w1s[j] + x1 * w1s[64 + j] + x2 * w1s[128 + j];
                float v  = agg[n * 16 + f] + ns * hw + b1[j];
                orow[j]  = (_Float16)leaky(v);
            }
        }
        __syncthreads();
    }
}

// ---------------------------------------------------------------------------
// Kernel 6: layer-2 dense GEMM  hw2[262144 x 64] = h1h @ W2  via WMMA.
// One 16x16 M-tile per wave, 8 waves/block -> 128 rows/block, 2048 blocks.
// A operand (16x32 f16, ISA 7.12.2): lanes 0-15 rows, halves 0-7 = K(hi*8+t),
// halves 8-15 = K(16+hi*8+t-8)  -> two 16B contiguous loads per K-half.
// B operand (32x16 f16): lanes 0-15 = N col, halves = K in the same order as A.
__global__ __launch_bounds__(256) void k_gemm2(
    const _Float16* __restrict__ h1h, const _Float16* __restrict__ w2h,
    float* __restrict__ hw2)
{
    const int wave = threadIdx.x >> 5;
    const int lane = threadIdx.x & 31;
    const int r    = lane & 15;
    const int hi   = lane >> 4;
    const int row0 = (blockIdx.x * 8 + wave) * 16;

    const _Float16* arow = h1h + (size_t)(row0 + r) * 64;
    AV a0, a1;
    a0.q[0] = *(const uint4*)(arow +  0 + hi * 8);
    a0.q[1] = *(const uint4*)(arow + 16 + hi * 8);
    a1.q[0] = *(const uint4*)(arow + 32 + hi * 8);
    a1.q[1] = *(const uint4*)(arow + 48 + hi * 8);

#pragma unroll
    for (int nt = 0; nt < 4; ++nt) {
        const _Float16* b0p = w2h + nt * 16 + r;
        const _Float16* b1p = w2h + (size_t)32 * 64 + nt * 16 + r;
        AV bk0, bk1;
#pragma unroll
        for (int s = 0; s < 16; ++s) {
            int k = (s < 8) ? (hi * 8 + s) : (16 + hi * 8 + (s - 8));
            bk0.v[s] = b0p[(size_t)k * 64];
            bk1.v[s] = b1p[(size_t)k * 64];
        }

        v8f c = {};
        c = __builtin_amdgcn_wmma_f32_16x16x32_f16(false, a0.v, false, bk0.v,
                                                   (short)0, c, false, false);
        c = __builtin_amdgcn_wmma_f32_16x16x32_f16(false, a1.v, false, bk1.v,
                                                   (short)0, c, false, false);
#pragma unroll
        for (int rr = 0; rr < 8; ++rr) {   // D: M = hi*8+rr, N = lane&15
            int m = hi * 8 + rr;
            hw2[(size_t)(row0 + m) * 64 + nt * 16 + r] = c[rr];
        }
    }
}

// ---------------------------------------------------------------------------
// Kernel 7: GCN layer 2 scatter + self + bias + LeakyReLU + mean-pool,
// fused per molecule. hw2 gathers hit L2 (256 KB/molecule << 192 MB L2).
__global__ __launch_bounds__(256) void k_gcn2(
    const int* __restrict__ ei, const float* __restrict__ b2,
    const float* __restrict__ nself, const float* __restrict__ norme,
    const float* __restrict__ hw2, _Float16* __restrict__ gh)
{
    __shared__ float agg[NATOM * 16];    // 64 KB
    __shared__ float gsum[16];

    const int b   = blockIdx.x;
    const int tid = threadIdx.x;
    const float* hb = hw2 + (size_t)b * NATOM * 64;

    for (int c = 0; c < 4; ++c) {
        for (int i = tid; i < NATOM * 16; i += 256) agg[i] = 0.0f;
        if (tid < 16) gsum[tid] = 0.0f;
        __syncthreads();

        for (int e = tid; e < NEDGE; e += 256) {
            int   s = ei[e];
            int   d = ei[NEDGE + e];
            float w = norme[e];
            const float4* hv = (const float4*)(hb + (size_t)s * 64 + c * 16);
#pragma unroll
            for (int q = 0; q < 4; ++q) {
                float4 v = hv[q];
                atomicAdd(&agg[d * 16 + q * 4 + 0], w * v.x);
                atomicAdd(&agg[d * 16 + q * 4 + 1], w * v.y);
                atomicAdd(&agg[d * 16 + q * 4 + 2], w * v.z);
                atomicAdd(&agg[d * 16 + q * 4 + 3], w * v.w);
            }
        }
        __syncthreads();

        float part[16];
#pragma unroll
        for (int f = 0; f < 16; ++f) part[f] = 0.0f;

        for (int n = tid; n < NATOM; n += 256) {
            float ns = nself[n];
            const float4* hv = (const float4*)(hb + (size_t)n * 64 + c * 16);
#pragma unroll
            for (int q = 0; q < 4; ++q) {
                float4 v = hv[q];
                float vv[4] = { v.x, v.y, v.z, v.w };
#pragma unroll
                for (int t = 0; t < 4; ++t) {
                    int f = q * 4 + t;
                    float val = agg[n * 16 + f] + ns * vv[t] + b2[c * 16 + f];
                    part[f] += leaky(val);
                }
            }
        }
#pragma unroll
        for (int f = 0; f < 16; ++f) atomicAdd(&gsum[f], part[f]);
        __syncthreads();

        if (tid < 16)
            gh[b * 64 + c * 16 + tid] = (_Float16)(gsum[tid] * (1.0f / NATOM));
        __syncthreads();
    }
}

// ---------------------------------------------------------------------------
// Kernel 8: projection  out[256x128] = leaky(g @ Wp + bp)  via WMMA.
// One block, 8 waves; each wave: 2 M-tiles x 8 N-tiles, K=64 (2 wmma each).
__global__ __launch_bounds__(256) void k_proj(
    const _Float16* __restrict__ gh, const _Float16* __restrict__ wph,
    const float* __restrict__ bp, float* __restrict__ out)
{
    const int wave = threadIdx.x >> 5;
    const int lane = threadIdx.x & 31;
    const int r    = lane & 15;
    const int hi   = lane >> 4;

    for (int mt = wave * 2; mt < wave * 2 + 2; ++mt) {
        const int row0 = mt * 16;
        const _Float16* arow = gh + (size_t)(row0 + r) * 64;
        AV a0, a1;
        a0.q[0] = *(const uint4*)(arow +  0 + hi * 8);
        a0.q[1] = *(const uint4*)(arow + 16 + hi * 8);
        a1.q[0] = *(const uint4*)(arow + 32 + hi * 8);
        a1.q[1] = *(const uint4*)(arow + 48 + hi * 8);

#pragma unroll
        for (int nt = 0; nt < 8; ++nt) {
            const _Float16* b0p = wph + nt * 16 + r;
            const _Float16* b1p = wph + (size_t)32 * 128 + nt * 16 + r;
            AV bk0, bk1;
#pragma unroll
            for (int s = 0; s < 16; ++s) {
                int k = (s < 8) ? (hi * 8 + s) : (16 + hi * 8 + (s - 8));
                bk0.v[s] = b0p[(size_t)k * 128];
                bk1.v[s] = b1p[(size_t)k * 128];
            }

            v8f c = {};
            c = __builtin_amdgcn_wmma_f32_16x16x32_f16(false, a0.v, false, bk0.v,
                                                       (short)0, c, false, false);
            c = __builtin_amdgcn_wmma_f32_16x16x32_f16(false, a1.v, false, bk1.v,
                                                       (short)0, c, false, false);
#pragma unroll
            for (int rr = 0; rr < 8; ++rr) {
                int m   = hi * 8 + rr;
                int col = nt * 16 + r;
                out[(size_t)(row0 + m) * 128 + col] = leaky(c[rr] + bp[col]);
            }
        }
    }
}

// ---------------------------------------------------------------------------
extern "C" void kernel_launch(void* const* d_in, const int* in_sizes, int n_in,
                              void* d_out, int out_size, void* d_ws, size_t ws_size,
                              hipStream_t stream) {
    const float* x  = (const float*)d_in[0];
    const int*   ei = (const int*)  d_in[1];   // [2, NEDGE]: row0=src, row1=dst
    const float* W1 = (const float*)d_in[2];
    const float* b1 = (const float*)d_in[3];
    const float* W2 = (const float*)d_in[4];
    const float* b2 = (const float*)d_in[5];
    const float* Wp = (const float*)d_in[6];
    const float* bp = (const float*)d_in[7];
    float* out = (float*)d_out;

    char* ws = (char*)d_ws;
    unsigned int* deg   = (unsigned int*)(ws + OFF_DEG);
    float*        dinv  = (float*)       (ws + OFF_DINV);
    float*        nself = (float*)       (ws + OFF_NSELF);
    float*        norme = (float*)       (ws + OFF_NORME);
    _Float16*     w2h   = (_Float16*)    (ws + OFF_W2H);
    _Float16*     wph   = (_Float16*)    (ws + OFF_WPH);
    _Float16*     gh    = (_Float16*)    (ws + OFF_GH);
    _Float16*     h1h   = (_Float16*)    (ws + OFF_H1H);
    float*        hw2   = (float*)       (ws + OFF_HW2);

    k_zero_deg   <<<NATOM / 256, 256, 0, stream>>>(deg);
    k_count_deg  <<<NEDGE / 256, 256, 0, stream>>>(ei, deg);
    k_norm_node  <<<NATOM / 256, 256, 0, stream>>>(deg, dinv, nself);
    k_norm_edge  <<<NEDGE / 256, 256, 0, stream>>>(ei, dinv, norme);
    k_cvt_weights<<<(64*64 + 64*128) / 256, 256, 0, stream>>>(W2, Wp, w2h, wph);

    k_gcn1 <<<BATCH, 256, 0, stream>>>(x, ei, W1, b1, nself, norme, h1h);
    k_gemm2<<<(BATCH * NATOM) / 128, 256, 0, stream>>>(h1h, w2h, hw2);
    k_gcn2 <<<BATCH, 256, 0, stream>>>(ei, b2, nself, norme, hw2, gh);
    k_proj <<<1, 256, 0, stream>>>(gh, wph, bp, out);
}
